// ResidualApproximateBNN_82214263980127
// MI455X (gfx1250) — matrix operand order repaired and checked by reading, not compiled, not run on hardware
//
#include <hip/hip_runtime.h>

// ---------------------------------------------------------------------------
// ResidualApproximateBNN forward on MI455X (gfx1250, wave32, WMMA).
// fp32 -> bf16 once; six GEMMs via v_wmma_f32_16x16x32_bf16 (f32 accum);
// branch-free fused bias + per-column activation epilogue; bf16 activations
// between layers; f32 final store.
// ---------------------------------------------------------------------------

typedef __bf16 bf16_t;
typedef __attribute__((ext_vector_type(16))) __bf16 v16bf;
typedef __attribute__((ext_vector_type(8)))  __bf16 v8bf;
typedef __attribute__((ext_vector_type(4)))  __bf16 v4bf;
typedef __attribute__((ext_vector_type(8)))  float  v8f;
typedef __attribute__((ext_vector_type(4)))  float  v4f;

#define TM 4   // 4 x 16 = 64 rows per wave
#define TN 4   // 4 x 16 = 64 cols per wave

// ---- fragment loaders (layouts per CDNA5 ISA 7.12.2) ----------------------
// A: 16x32 bf16, lane L holds row (L&15); lanes 0-15 carry K {0..7,16..23},
// lanes 16-31 carry K {8..15,24..31}. Two contiguous 8-element runs.
__device__ __forceinline__ v16bf load_a_frag(const bf16_t* __restrict__ A,
                                             int lda, int row0, int k0) {
  const int lane = threadIdx.x & 31;
  const int r = lane & 15;
  const int h = lane >> 4;
  const bf16_t* p = A + (size_t)(row0 + r) * (size_t)lda + (size_t)(k0 + 8 * h);
  union { v16bf v; v8bf h2[2]; } u;
  u.h2[0] = *(const v8bf*)(p);        // K = k0 + 8h + [0..7]
  u.h2[1] = *(const v8bf*)(p + 16);   // K = k0 + 16 + 8h + [0..7]
  return u.v;
}

// B: 32x16 bf16 (K x N), lane L holds column (L&15); lanes 0-15 carry
// K {0..7,16..23}, lanes 16-31 carry K {8..15,24..31} (same order as A).
// W is [N,K] row-major, so a column of B == a row of W: contiguous.
__device__ __forceinline__ v16bf load_b_frag(const bf16_t* __restrict__ W,
                                             int ldw, int col0, int k0) {
  const int lane = threadIdx.x & 31;
  const int c = lane & 15;
  const int h = lane >> 4;
  const bf16_t* p = W + (size_t)(col0 + c) * (size_t)ldw + (size_t)(k0 + 8 * h);
  union { v16bf v; v8bf h2[2]; } u;
  u.h2[0] = *(const v8bf*)(p);
  u.h2[1] = *(const v8bf*)(p + 16);
  return u.v;
}

// ---- branch-free activation selector --------------------------------------
// All five candidates are computed with branchless math and chosen with
// v_cndmask chains; no exec-mask manipulation in the (fully unrolled) epilogue.
__device__ __forceinline__ float apply_act(float x, bool is0, bool is1,
                                           bool is2, bool is3) {
  const float xneg = fminf(x, 0.f);
  const float xpos = fmaxf(x, 0.f);
  const float relu  = xpos;                              // id 0
  const float sig   = 1.f / (1.f + __expf(-x));          // id 1
  const float e2x   = __expf(2.f * x);                   // tanh, exact at +-inf
  const float th    = 1.f - 2.f / (e2x + 1.f);           // id 2
  const float leaky = xpos + 0.1f * xneg;                // id 3
  const float sc = 1.0507009873554805f;
  const float al = 1.6732632423543772f;
  const float se = sc * (xpos + al * (__expf(xneg) - 1.f)); // id 4 (SELU)
  float y = se;
  y = is3 ? leaky : y;
  y = is2 ? th    : y;
  y = is1 ? sig   : y;
  y = is0 ? relu  : y;
  return y;
}

// ---- fp32 -> bf16 conversion (vectorized, grid-stride) --------------------
__global__ __launch_bounds__(256) void f32_to_bf16_kernel(
    const float* __restrict__ src, bf16_t* __restrict__ dst, long n4) {
  long i = (long)blockIdx.x * blockDim.x + threadIdx.x;
  const long stride = (long)gridDim.x * blockDim.x;
  for (; i < n4; i += stride) {
    const v4f x = *(const v4f*)(src + 4 * i);
    v4bf y;
    y.x = (bf16_t)x.x; y.y = (bf16_t)x.y;
    y.z = (bf16_t)x.z; y.w = (bf16_t)x.w;
    *(v4bf*)(dst + 4 * i) = y;
  }
}

// ---- fused GEMM + bias + per-column activation ----------------------------
// C[M,N] = act( A[M,K] @ W[N,K]^T + b ), per-wave 64x64 tile, K-step 32.
__global__ __launch_bounds__(256) void gemm_bias_act_kernel(
    const bf16_t* __restrict__ A, int lda,
    const bf16_t* __restrict__ W,            // [N,K] row-major, ldw == K
    const float*  __restrict__ bias,
    const int*    __restrict__ act,
    int M, int N, int K,
    bf16_t* __restrict__ out1, int ldo1,     // primary bf16 store (nullable)
    bf16_t* __restrict__ out2, int ldo2,     // secondary bf16 store (nullable)
    float*  __restrict__ outf, int ldof) {   // f32 store (nullable)
  const int wave    = (int)((blockIdx.x * blockDim.x + threadIdx.x) >> 5);
  const int tiles_n = N >> 6;
  const int tiles_m = M >> 6;
  if (wave >= tiles_m * tiles_n) return;     // wave-uniform: EXEC stays all-1s

  const int tm   = wave / tiles_n;
  const int tn   = wave - tm * tiles_n;
  const int row0 = tm << 6;
  const int col0 = tn << 6;

  v8f acc[TM][TN] = {};

  for (int k0 = 0; k0 < K; k0 += 32) {
    v16bf a[TM], b[TN];
#pragma unroll
    for (int i = 0; i < TM; ++i) a[i] = load_a_frag(A, lda, row0 + 16 * i, k0);
#pragma unroll
    for (int j = 0; j < TN; ++j) b[j] = load_b_frag(W, K, col0 + 16 * j, k0);
#pragma unroll
    for (int i = 0; i < TM; ++i)
#pragma unroll
      for (int j = 0; j < TN; ++j)
        acc[i][j] = __builtin_amdgcn_wmma_f32_16x16x32_bf16(
            /*neg_a=*/false, a[i], /*neg_b=*/false, b[j],
            /*c_mod=*/(short)0, acc[i][j],
            /*reuse_a=*/false, /*reuse_b=*/false);
  }

  // Epilogue. C/D layout: lane L, VGPR r -> row = 8*(L>>4)+r, col = L&15.
  const int lane = threadIdx.x & 31;
  const int cl   = lane & 15;
  const int rh   = (lane >> 4) * 8;

#pragma unroll
  for (int j = 0; j < TN; ++j) {
    const int   col = col0 + 16 * j + cl;
    const float bv  = bias[col];
    const int   id  = act[col];
    const bool  is0 = (id == 0), is1 = (id == 1);
    const bool  is2 = (id == 2), is3 = (id == 3);
#pragma unroll
    for (int i = 0; i < TM; ++i) {
      const int rowb = row0 + 16 * i + rh;
#pragma unroll
      for (int r = 0; r < 8; ++r) {
        const float y   = apply_act(acc[i][j][r] + bv, is0, is1, is2, is3);
        const int   row = rowb + r;
        if (out1) out1[(size_t)row * ldo1 + col] = (bf16_t)y;
        if (out2) out2[(size_t)row * ldo2 + col] = (bf16_t)y;
        if (outf) outf[(size_t)row * ldof + col] = y;
      }
    }
  }
}

// ---------------------------------------------------------------------------
// Host-side orchestration
// ---------------------------------------------------------------------------
static inline void convert(const float* s, bf16_t* d, long n, hipStream_t st) {
  f32_to_bf16_kernel<<<1024, 256, 0, st>>>(s, d, n / 4);
}

static inline void gemm(const bf16_t* A, int lda, const bf16_t* W,
                        const float* bias, const int* act,
                        int M, int N, int K,
                        bf16_t* o1, int l1, bf16_t* o2, int l2,
                        float* of, int lf, hipStream_t st) {
  const int waves  = (M >> 6) * (N >> 6);
  const int blocks = (waves + 7) / 8;        // 8 waves (256 threads) per block
  gemm_bias_act_kernel<<<blocks, 256, 0, st>>>(A, lda, W, bias, act, M, N, K,
                                               o1, l1, o2, l2, of, lf);
}

extern "C" void kernel_launch(void* const* d_in, const int* in_sizes, int n_in,
                              void* d_out, int out_size, void* d_ws, size_t ws_size,
                              hipStream_t stream) {
  (void)in_sizes; (void)n_in; (void)out_size;

  constexpr int M   = 64 * 128;   // 8192 rows
  constexpr int IN  = 512;
  constexpr int X   = 2048;
  constexpr int OUT = 512;

  const float* x_f32  = (const float*)d_in[0];
  const float* W_in   = (const float*)d_in[1];
  const float* b_in   = (const float*)d_in[2];
  const float* W_h0   = (const float*)d_in[3];
  const float* b_h0   = (const float*)d_in[4];
  const float* W_h1   = (const float*)d_in[5];
  const float* b_h1   = (const float*)d_in[6];
  const float* W_h2   = (const float*)d_in[7];
  const float* b_h2   = (const float*)d_in[8];
  const float* W_h3   = (const float*)d_in[9];
  const float* b_h3   = (const float*)d_in[10];
  const float* W_out  = (const float*)d_in[11];
  const float* b_out  = (const float*)d_in[12];
  const int*   act_in = (const int*)d_in[13];
  const int*   act_h  = (const int*)d_in[14];   // [4, X]
  const int*   act_o  = (const int*)d_in[15];
  float*       out    = (float*)d_out;

  // Workspace layout (bf16 elements; every block is 256B aligned by size).
  bf16_t* ws = (bf16_t*)d_ws;
  size_t o = 0;
  bf16_t* wb_in  = ws + o; o += (size_t)X * IN;        // 2048x512
  bf16_t* wb_h0  = ws + o; o += (size_t)X * X;         // 2048x2048
  bf16_t* wb_h1  = ws + o; o += (size_t)X * X;
  bf16_t* wb_h2  = ws + o; o += (size_t)X * 2 * X;     // 2048x4096
  bf16_t* wb_h3  = ws + o; o += (size_t)X * X;
  bf16_t* wb_out = ws + o; o += (size_t)OUT * X;       // 512x2048
  bf16_t* xb     = ws + o; o += (size_t)M * IN;        // 8192x512
  bf16_t* bufA   = ws + o; o += (size_t)M * X;         // h0, later h3
  bf16_t* bufB   = ws + o; o += (size_t)M * X;         // h1, later h4
  bf16_t* cat    = ws + o; o += (size_t)M * 2 * X;     // [h2 | h1], ld 4096
  if (ws_size < o * sizeof(bf16_t)) return;            // workspace too small

  // 1) fp32 -> bf16 conversions (weights + input), one cheap pass.
  convert(W_in,  wb_in,  (long)X * IN,    stream);
  convert(W_h0,  wb_h0,  (long)X * X,     stream);
  convert(W_h1,  wb_h1,  (long)X * X,     stream);
  convert(W_h2,  wb_h2,  (long)X * 2 * X, stream);
  convert(W_h3,  wb_h3,  (long)X * X,     stream);
  convert(W_out, wb_out, (long)OUT * X,   stream);
  convert(x_f32, xb,     (long)M * IN,    stream);

  // 2) Layer chain (all on `stream`, implicitly ordered).
  // h0 = act_in(x @ W_in^T + b_in)
  gemm(xb, IN, wb_in, b_in, act_in, M, X, IN,
       bufA, X, nullptr, 0, nullptr, 0, stream);
  // h1 = act_h[0](h0 @ W_h0^T + b_h0); dual-store into cat[:,2048:4096]
  gemm(bufA, X, wb_h0, b_h0, act_h + 0 * X, M, X, X,
       bufB, X, cat + X, 2 * X, nullptr, 0, stream);
  // h2 = act_h[1](h1 @ W_h1^T + b_h1) -> cat[:,0:2048]
  gemm(bufB, X, wb_h1, b_h1, act_h + 1 * X, M, X, X,
       cat, 2 * X, nullptr, 0, nullptr, 0, stream);
  // h3 = act_h[2](concat(h2,h1) @ W_h2^T + b_h2)   (K = 4096)
  gemm(cat, 2 * X, wb_h2, b_h2, act_h + 2 * X, M, X, 2 * X,
       bufA, X, nullptr, 0, nullptr, 0, stream);
  // h4 = act_h[3](h3 @ W_h3^T + b_h3)
  gemm(bufA, X, wb_h3, b_h3, act_h + 3 * X, M, X, X,
       bufB, X, nullptr, 0, nullptr, 0, stream);
  // out = act_out(h4 @ W_out^T + b_out), f32 store
  gemm(bufB, X, wb_out, b_out, act_o, M, OUT, X,
       nullptr, 0, nullptr, 0, out, OUT, stream);
}
